// MambaModel_37082747633864
// MI455X (gfx1250) — compile-verified, matrix operand order checked by reading, not run on hardware
//
#include <hip/hip_runtime.h>
#include <hip/hip_bf16.h>
#include <math.h>

// Model constants (match reference)
#define MTOT   4096      // B*L
#define DMODEL 256
#define NLAYER 4
#define NSTATE 16
#define EDIM   512
#define KCONV  4
#define DTR    16
#define VOCAB  32000
#define LSEQ   2048
#define DBCW   48        // DTR + 2*NSTATE

typedef __attribute__((ext_vector_type(16))) _Float16 v16h;
typedef __attribute__((ext_vector_type(8)))  _Float16 v8h;
typedef __attribute__((ext_vector_type(8)))  float    v8f;

// fast sigmoid / silu using hardware v_exp_f32 + v_rcp_f32
__device__ __forceinline__ float fast_sigmoid(float x) {
  return __builtin_amdgcn_rcpf(1.f + __expf(-x));
}

// ---------------- elementwise helpers ----------------

__global__ void f32_to_f16_kernel(const float* __restrict__ src,
                                  _Float16* __restrict__ dst, int n) {
  int i = blockIdx.x * blockDim.x + threadIdx.x;
  if (i < n) dst[i] = (_Float16)src[i];
}

__global__ void embed_kernel(const int* __restrict__ x, const float* __restrict__ et,
                             float* __restrict__ h) {
  int i = blockIdx.x * blockDim.x + threadIdx.x;   // over MTOT*DMODEL
  int m = i >> 8, d = i & 255;
  h[i] = et[(size_t)x[m] * DMODEL + d];
}

__global__ void rmsnorm_kernel(const float* __restrict__ h, const float* __restrict__ w,
                               _Float16* __restrict__ out) {
  __shared__ float red[DMODEL];
  int row = blockIdx.x, d = threadIdx.x;
  float v = h[(size_t)row * DMODEL + d];
  red[d] = v * v;
  __syncthreads();
  for (int s = DMODEL / 2; s > 0; s >>= 1) {
    if (d < s) red[d] += red[d + s];
    __syncthreads();
  }
  float r = rsqrtf(red[0] / (float)DMODEL + 1e-5f);
  out[(size_t)row * DMODEL + d] = (_Float16)(v * r * w[d]);
}

__global__ void layernorm_kernel(const float* __restrict__ h, const float* __restrict__ w,
                                 const float* __restrict__ b, _Float16* __restrict__ out) {
  __shared__ float s1[DMODEL], s2[DMODEL];
  int row = blockIdx.x, d = threadIdx.x;
  float v = h[(size_t)row * DMODEL + d];
  s1[d] = v; s2[d] = v * v;
  __syncthreads();
  for (int s = DMODEL / 2; s > 0; s >>= 1) {
    if (d < s) { s1[d] += s1[d + s]; s2[d] += s2[d + s]; }
    __syncthreads();
  }
  float mu  = s1[0] / (float)DMODEL;
  float var = s2[0] / (float)DMODEL - mu * mu;
  out[(size_t)row * DMODEL + d] =
      (_Float16)((v - mu) * rsqrtf(var + 1e-5f) * w[d] + b[d]);
}

// causal depthwise conv (K=4) + SiLU; xz row stride = 2*EDIM, conv reads first EDIM cols
__global__ void conv_silu_kernel(const float* __restrict__ xz, const float* __restrict__ cw,
                                 const float* __restrict__ cb, float* __restrict__ xc,
                                 _Float16* __restrict__ xc16) {
  int i = blockIdx.x * blockDim.x + threadIdx.x;   // over MTOT*EDIM
  int m = i >> 9, e = i & (EDIM - 1);
  int b = m >> 11, l = m & (LSEQ - 1);
  float acc = cb[e];
#pragma unroll
  for (int t = 0; t < KCONV; ++t) {
    int lp = l + t - (KCONV - 1);
    if (lp >= 0)
      acc = fmaf(xz[((size_t)((b << 11) + lp)) * (2 * EDIM) + e], cw[e * KCONV + t], acc);
  }
  float s = acc * fast_sigmoid(acc);               // SiLU
  xc[i] = s;
  xc16[i] = (_Float16)s;
}

// y * silu(z) -> f16 for out_proj A operand
__global__ void gate_kernel(const float* __restrict__ y, const float* __restrict__ xz,
                            _Float16* __restrict__ yg) {
  int i = blockIdx.x * blockDim.x + threadIdx.x;   // over MTOT*EDIM
  int m = i >> 9, e = i & (EDIM - 1);
  float z = xz[(size_t)m * (2 * EDIM) + EDIM + e];
  yg[i] = (_Float16)(y[i] * z * fast_sigmoid(z));
}

// selective scan: 16 lanes = 16 states of one (b,e) channel; 2 channels per wave32
__global__ void scan_kernel(const float* __restrict__ delta, const float* __restrict__ xc,
                            const float* __restrict__ dbc, const float* __restrict__ A_log,
                            const float* __restrict__ Dp, float* __restrict__ y) {
  int grp = (blockIdx.x * blockDim.x + threadIdx.x) >> 4;  // 0 .. B*EDIM-1
  int n   = threadIdx.x & 15;
  int b   = grp >> 9, e = grp & (EDIM - 1);
  float An    = -__expf(A_log[e * NSTATE + n]);
  float dpe   = Dp[e];
  float state = 0.f;
  for (int l = 0; l < LSEQ; ++l) {
    int m = (b << 11) + l;
    float dv = delta[(size_t)m * EDIM + e];
    float xv = xc[(size_t)m * EDIM + e];
    float Bn = dbc[(size_t)m * DBCW + DTR + n];
    float Cn = dbc[(size_t)m * DBCW + DTR + NSTATE + n];
    float dA = __expf(dv * An);
    state = fmaf(dA, state, dv * Bn * xv);
    float acc = state * Cn;
    acc += __shfl_xor(acc, 1, 16);
    acc += __shfl_xor(acc, 2, 16);
    acc += __shfl_xor(acc, 4, 16);
    acc += __shfl_xor(acc, 8, 16);
    if (n == 0) y[(size_t)m * EDIM + e] = acc + dpe * xv;
  }
}

// ---------------- tiled WMMA GEMM: C[M,N] = A[M,K](f16) * W[N,K]^T (f16) ----------
// Block = 256 threads (8 waves). Block tile: 32 rows x 512 cols.
// Wave tile: 32x64 = 2 M-tiles x 4 N-tiles -> 8 wmma per K-step from
// 4 ds_load_b128 (A from LDS) + 8 global_load_b128 (B fragments).
// A panel (32 x K) staged in LDS once per block via GLOBAL_LOAD_ASYNC_TO_LDS_B128
// (ASYNCcnt), rows padded by 8 halfs so fragment reads are bank-conflict free.
// gfx1250 16-bit fragment mapping: element e of v16h holds
// K = 16*(e/8) + 8*half + (e%8), half = lane/16; D/C: c[r] -> (m=r+8*half, n=lane&15).
template <int K, bool BIAS, bool RESID, bool STF16>
__global__ __launch_bounds__(256) void wmma_gemm_tiled_kernel(
    const _Float16* __restrict__ A, int lda,
    const _Float16* __restrict__ Bw, int ldb,
    float* __restrict__ C, int ldc,
    _Float16* __restrict__ C16,
    const float* __restrict__ bias, int N) {
  constexpr int LDS_STRIDE = K + 8;                 // halfs; +16B pad per row
  __shared__ _Float16 As[32 * LDS_STRIDE];

  const int tid    = threadIdx.x;
  const int mblock = blockIdx.y * 32;

  // ---- stage A panel (32 x K) into LDS with async global->LDS copies ----
  {
    constexpr int CPR = K / 8;                      // 16B chunks per row
    uint32_t lbase = (uint32_t)(uintptr_t)As;       // low 32 bits = LDS byte address
#pragma unroll 2
    for (int c = tid; c < 32 * CPR; c += 256) {
      int row = c / CPR, col = (c % CPR) * 8;
      unsigned long long g =
          (unsigned long long)(A + (size_t)(mblock + row) * lda + col);
      uint32_t l = lbase + (uint32_t)(row * LDS_STRIDE + col) * 2;
      asm volatile("global_load_async_to_lds_b128 %0, %1, off"
                   :: "v"(l), "v"(g) : "memory");
    }
    asm volatile("s_wait_asynccnt 0" ::: "memory");
  }
  __syncthreads();

  const int wave  = tid >> 5;
  const int lane  = tid & 31;
  const int half  = lane >> 4;
  const int lm    = lane & 15;
  const int nwave = blockIdx.x * 512 + wave * 64;   // wave's first output column
  if (nwave >= N) return;                           // wave-uniform

  v8f c[2][4];
#pragma unroll
  for (int i = 0; i < 2; ++i)
#pragma unroll
    for (int j = 0; j < 4; ++j) c[i][j] = (v8f){};

  for (int ko = 0; ko < K; ko += 32) {
    v16h a[2];
#pragma unroll
    for (int i = 0; i < 2; ++i) {
      const _Float16* ap = &As[(i * 16 + lm) * LDS_STRIDE + ko + 8 * half];
      v8h a0 = *(const v8h*)ap;
      v8h a1 = *(const v8h*)(ap + 16);
#pragma unroll
      for (int e = 0; e < 8; ++e) { a[i][e] = a0[e]; a[i][e + 8] = a1[e]; }
    }
#pragma unroll
    for (int j = 0; j < 4; ++j) {
      int ng = nwave + j * 16 + lm;
      if (ng > N - 1) ng = N - 1;                   // clamp: loaded but never stored
      const _Float16* bp = Bw + (size_t)ng * ldb + ko + 8 * half;
      v8h b0 = *(const v8h*)bp;
      v8h b1 = *(const v8h*)(bp + 16);
      v16h b;
#pragma unroll
      for (int e = 0; e < 8; ++e) { b[e] = b0[e]; b[e + 8] = b1[e]; }
      c[0][j] = __builtin_amdgcn_wmma_f32_16x16x32_f16(false, a[0], false, b,
                                                       (short)0, c[0][j], false, false);
      c[1][j] = __builtin_amdgcn_wmma_f32_16x16x32_f16(false, a[1], false, b,
                                                       (short)0, c[1][j], false, false);
    }
  }

#pragma unroll
  for (int j = 0; j < 4; ++j) {
    int ncol = nwave + j * 16;
    if (ncol >= N) break;                           // uniform per wave
#pragma unroll
    for (int i = 0; i < 2; ++i) {
#pragma unroll
      for (int r = 0; r < 8; ++r) {
        int mo = mblock + i * 16 + r + 8 * half;
        int no = ncol + lm;
        float v = c[i][j][r];
        if (BIAS)  v += bias[no];
        if (RESID) v += C[(size_t)mo * ldc + no];
        C[(size_t)mo * ldc + no] = v;
        if (STF16) C16[(size_t)mo * ldc + no] = (_Float16)v;
      }
    }
  }
}

// ---------------- dt_proj: (M,16) x (512,16)^T + bias, softplus -> delta --------
// K=16 < 32: element-guarded fragments, zero-filled for k >= 16. One wmma per tile.
__global__ __launch_bounds__(128) void dt_wmma_kernel(
    const _Float16* __restrict__ A,   // dbc16, lda = DBCW, delta part = cols 0..15
    const _Float16* __restrict__ Bw,  // (EDIM,16)
    float* __restrict__ C,            // delta, ldc = EDIM
    const float* __restrict__ bias) {
  int wave  = threadIdx.x >> 5;
  int lane  = threadIdx.x & 31;
  int ntile = blockIdx.x * 4 + wave;
  int mtile = blockIdx.y;
  int half  = lane >> 4;
  int lm    = lane & 15;
  int mg    = mtile * 16 + lm;
  int ng    = ntile * 16 + lm;
  v16h a, b;
#pragma unroll
  for (int e = 0; e < 16; ++e) {
    int k = ((e >> 3) << 4) + 8 * half + (e & 7);
    a[e] = (k < DTR) ? A[(size_t)mg * DBCW + k] : (_Float16)0.f;
    b[e] = (k < DTR) ? Bw[(size_t)ng * DTR + k] : (_Float16)0.f;
  }
  v8f c = {};
  c = __builtin_amdgcn_wmma_f32_16x16x32_f16(false, a, false, b, (short)0, c,
                                             false, false);
#pragma unroll
  for (int r = 0; r < 8; ++r) {
    int mo = mtile * 16 + r + 8 * half;
    int no = ntile * 16 + lm;
    float v = c[r] + bias[no];
    // fast softplus: hardware v_exp_f32 + v_log_f32 (inputs are small; exact for v>20)
    v = (v > 20.f) ? v : __logf(1.f + __expf(v));
    C[(size_t)mo * EDIM + no] = v;
  }
}

// ---------------- host side ----------------

extern "C" void kernel_launch(void* const* d_in, const int* in_sizes, int n_in,
                              void* d_out, int out_size, void* d_ws, size_t ws_size,
                              hipStream_t stream) {
  (void)in_sizes; (void)n_in; (void)out_size; (void)ws_size;
  const int*   x        = (const int*)  d_in[0];
  const float* embed_t  = (const float*)d_in[1];
  const float* rms_w    = (const float*)d_in[2];   // NL*D
  const float* in_pw    = (const float*)d_in[3];   // NL*1024*256
  const float* conv_w   = (const float*)d_in[4];   // NL*512*4
  const float* conv_b   = (const float*)d_in[5];   // NL*512
  const float* x_pw     = (const float*)d_in[6];   // NL*48*512
  const float* dt_pw    = (const float*)d_in[7];   // NL*512*16
  const float* dt_pb    = (const float*)d_in[8];   // NL*512
  const float* A_log    = (const float*)d_in[9];   // NL*512*16
  const float* D_p      = (const float*)d_in[10];  // NL*512
  const float* out_pw   = (const float*)d_in[11];  // NL*256*512
  const float* norm_w   = (const float*)d_in[12];
  const float* norm_b   = (const float*)d_in[13];
  const float* head_w   = (const float*)d_in[14];  // V*256
  const float* head_b   = (const float*)d_in[15];
  float* out = (float*)d_out;                      // MTOT*V fp32

  // workspace carve-out
  char* p = (char*)d_ws;
  auto take = [&](size_t bytes) -> char* {
    char* r = p;
    p += (bytes + 255) & ~(size_t)255;
    return r;
  };
  float*    h     = (float*)   take((size_t)MTOT * DMODEL * 4);
  _Float16* xin16 = (_Float16*)take((size_t)MTOT * DMODEL * 2);
  float*    xz    = (float*)   take((size_t)MTOT * 2 * EDIM * 4);
  float*    xc    = (float*)   take((size_t)MTOT * EDIM * 4);
  _Float16* xc16  = (_Float16*)take((size_t)MTOT * EDIM * 2);
  float*    dbc   = (float*)   take((size_t)MTOT * DBCW * 4);
  _Float16* dbc16 = (_Float16*)take((size_t)MTOT * DBCW * 2);
  float*    delta = (float*)   take((size_t)MTOT * EDIM * 4);
  float*    yb    = (float*)   take((size_t)MTOT * EDIM * 4);
  _Float16* yg16  = (_Float16*)take((size_t)MTOT * EDIM * 2);
  _Float16* hn16  = (_Float16*)take((size_t)MTOT * DMODEL * 2);
  _Float16* w_in  = (_Float16*)take((size_t)NLAYER * 2 * EDIM * DMODEL * 2);
  _Float16* w_xp  = (_Float16*)take((size_t)NLAYER * DBCW * EDIM * 2);
  _Float16* w_dt  = (_Float16*)take((size_t)NLAYER * EDIM * DTR * 2);
  _Float16* w_op  = (_Float16*)take((size_t)NLAYER * DMODEL * EDIM * 2);
  _Float16* w_hd  = (_Float16*)take((size_t)VOCAB * DMODEL * 2);

  auto cvt = [&](const float* src, _Float16* dst, int n) {
    f32_to_f16_kernel<<<(n + 255) / 256, 256, 0, stream>>>(src, dst, n);
  };
  cvt(in_pw,  w_in, NLAYER * 2 * EDIM * DMODEL);
  cvt(x_pw,   w_xp, NLAYER * DBCW * EDIM);
  cvt(dt_pw,  w_dt, NLAYER * EDIM * DTR);
  cvt(out_pw, w_op, NLAYER * DMODEL * EDIM);
  cvt(head_w, w_hd, VOCAB * DMODEL);

  // embed gather
  embed_kernel<<<(MTOT * DMODEL) / 256, 256, 0, stream>>>(x, embed_t, h);

  const int MB = MTOT / 32;             // 128 M-blocks of 32 rows
  for (int i = 0; i < NLAYER; ++i) {
    rmsnorm_kernel<<<MTOT, DMODEL, 0, stream>>>(h, rms_w + (size_t)i * DMODEL, xin16);

    // in_proj: (M,256) x (1024,256)^T -> xz
    wmma_gemm_tiled_kernel<DMODEL, false, false, false>
        <<<dim3((2 * EDIM + 511) / 512, MB), 256, 0, stream>>>(
            xin16, DMODEL, w_in + (size_t)i * 2 * EDIM * DMODEL, DMODEL,
            xz, 2 * EDIM, nullptr, nullptr, 2 * EDIM);

    conv_silu_kernel<<<(MTOT * EDIM) / 256, 256, 0, stream>>>(
        xz, conv_w + (size_t)i * EDIM * KCONV, conv_b + (size_t)i * EDIM, xc, xc16);

    // x_proj: (M,512) x (48,512)^T -> dbc (f32 + f16 mirror)
    wmma_gemm_tiled_kernel<EDIM, false, false, true>
        <<<dim3((DBCW + 511) / 512, MB), 256, 0, stream>>>(
            xc16, EDIM, w_xp + (size_t)i * DBCW * EDIM, EDIM,
            dbc, DBCW, dbc16, nullptr, DBCW);

    // dt_proj: (M,16) x (512,16)^T + bias, softplus -> delta
    dt_wmma_kernel<<<dim3(EDIM / 64, MTOT / 16), 128, 0, stream>>>(
        dbc16, w_dt + (size_t)i * EDIM * DTR, delta, dt_pb + (size_t)i * EDIM);

    // selective scan
    scan_kernel<<<(2 * EDIM * 16) / 256, 256, 0, stream>>>(
        delta, xc, dbc, A_log + (size_t)i * EDIM * NSTATE, D_p + (size_t)i * EDIM, yb);

    gate_kernel<<<(MTOT * EDIM) / 256, 256, 0, stream>>>(yb, xz, yg16);

    // out_proj with fused residual: h += yg @ Wout^T
    wmma_gemm_tiled_kernel<EDIM, false, true, false>
        <<<dim3((DMODEL + 511) / 512, MB), 256, 0, stream>>>(
            yg16, EDIM, w_op + (size_t)i * DMODEL * EDIM, EDIM,
            h, DMODEL, nullptr, nullptr, DMODEL);
  }

  layernorm_kernel<<<MTOT, DMODEL, 0, stream>>>(h, norm_w, norm_b, hn16);

  // head: (M,256) x (32000,256)^T + bias -> out
  wmma_gemm_tiled_kernel<DMODEL, true, false, false>
      <<<dim3((VOCAB + 511) / 512, MB), 256, 0, stream>>>(
          hn16, DMODEL, w_hd, DMODEL, out, VOCAB, nullptr, head_b, VOCAB);
}
